// ModulatedConv1d_29953101922875
// MI455X (gfx1250) — matrix operand order repaired and checked by reading, not compiled, not run on hardware
//
#include <hip/hip_runtime.h>

typedef _Float16 v16h __attribute__((ext_vector_type(16)));
typedef _Float16 v8h  __attribute__((ext_vector_type(8)));
typedef float    v8f  __attribute__((ext_vector_type(8)));

#define BB   16
#define II   256
#define OO   256
#define TT   512
#define LL   8192
#define KW   7
#define PADW 3

#define LTILE 64
#define KTILE 32
#define XSTR  40   // Xs row stride in halves
#define SRW   73   // Sraw row stride in floats (odd -> conflict-free strided reads)

// ---------------- Kernel 1: mod = t @ mod_w^T + mod_b  (B x I) ----------------
__global__ __launch_bounds__(256)
void mod_kernel(const float* __restrict__ t, const float* __restrict__ mod_w,
                const float* __restrict__ mod_b, float* __restrict__ mod) {
    __shared__ float ts[TT];
    const int b = blockIdx.x;
    const int i = threadIdx.x;
    ts[i]       = t[b * TT + i];
    ts[i + 256] = t[b * TT + i + 256];
    __syncthreads();
    float acc = mod_b[i];
    const float* wr = mod_w + (size_t)i * TT;
    #pragma unroll 8
    for (int k = 0; k < TT; ++k) acc = fmaf(ts[k], wr[k], acc);
    mod[b * II + i] = acc;
}

// ------- Kernel 2: bw[b,o,i] = f16( demod * proj_w[o,i]*(mod[b,i]+1) ) --------
__global__ __launch_bounds__(256)
void bw_kernel(const float* __restrict__ proj_w, const float* __restrict__ mod,
               _Float16* __restrict__ bw) {
    __shared__ float red[256];
    const int bo = blockIdx.x;          // b*O + o
    const int b  = bo >> 8;
    const int o  = bo & 255;
    const int i  = threadIdx.x;         // I == 256 exactly
    const float v = proj_w[(size_t)o * II + i] * (mod[b * II + i] + 1.0f);
    red[i] = v * v;
    __syncthreads();
    #pragma unroll
    for (int s = 128; s > 0; s >>= 1) {
        if (i < s) red[i] += red[i + s];
        __syncthreads();
    }
    const float demod = rsqrtf(red[0] + 1e-8f);
    bw[(size_t)bo * II + i] = (_Float16)(v * demod);
}

// ------- Kernel 3: fused depthwise-conv + per-batch GEMM via WMMA f16 --------
// out[b] (256 x 8192) = bw[b] (256 x 256) @ conv(bx[b]) (256 x 8192) + proj_b
__global__ __launch_bounds__(256)
void conv_gemm_kernel(const float* __restrict__ bx, const float* __restrict__ conv_w,
                      const float* __restrict__ proj_b, const _Float16* __restrict__ bw,
                      float* __restrict__ out) {
    __shared__ float    Sraw[KTILE][SRW];   // 32 x 73 f32 raw bx slab (cols 0..71 written)
    __shared__ _Float16 Xs[LTILE][XSTR];    // 64 x 40 f16, transposed conv output
    __shared__ float    cs[KW][II];         // conv weights, k-major
    __shared__ float    pbs[OO];            // proj bias

    const int b    = blockIdx.x >> 7;       // 128 L-tiles per batch
    const int l0   = (blockIdx.x & 127) * LTILE;
    const int tid  = threadIdx.x;
    const int wave = tid >> 5;
    const int lane = tid & 31;

    // one-time staging of conv weights (transposed) and bias
    for (int idx = tid; idx < KW * II; idx += 256) {
        const int k = idx >> 8, i = idx & 255;
        cs[k][i] = conv_w[i * KW + k];
    }
    pbs[tid] = proj_b[tid];

    v8f acc[2][4];
    const v8f vzero = {0.f, 0.f, 0.f, 0.f, 0.f, 0.f, 0.f, 0.f};
    #pragma unroll
    for (int mt = 0; mt < 2; ++mt)
        #pragma unroll
        for (int nt = 0; nt < 4; ++nt) acc[mt][nt] = vzero;

    const float*    bxb = bx + (size_t)b * II * LL;
    const _Float16* bwb = bw + (size_t)b * OO * II;

    // fixed staging pattern (no div/mod, no load branches)
    const int c    = tid & 63;              // main columns 0..63
    const int r0   = tid >> 6;              // 0..3 -> rows r0*8+j
    const int glm  = l0 - PADW + c;
    const int glmc = min(max(glm, 0), LL - 1);
    const bool okm = (glm == glmc);
    const int hr   = tid >> 3;              // halo: 32 rows x 8 cols (64..71)
    const int hc   = tid & 7;
    const int glh  = l0 - PADW + 64 + hc;
    const int glhc = min(max(glh, 0), LL - 1);
    const bool okh = (glh == glhc);

    const int mrow  = lane & 15;
    const int khalf = (lane >> 4) * 8;      // A: lanes 16-31 start at K=8
    const int kbb   = (lane >> 4) * 8;      // B: lanes 16-31 start at K=8

    for (int kk = 0; kk < II; kk += KTILE) {
        const float* bxk = bxb + (size_t)kk * LL;

        // ---- stage raw bx slab: coalesced main body + padded halo, branchless ----
        #pragma unroll
        for (int j = 0; j < 8; ++j) {
            const int r = r0 * 8 + j;       // rows 0..31
            const float v = bxk[(size_t)r * LL + glmc];
            Sraw[r][c] = okm ? v : 0.0f;
        }
        {
            const float v = bxk[(size_t)hr * LL + glhc];
            Sraw[hr][64 + hc] = okh ? v : 0.0f;   // cols 70,71 staged but never read
        }

        // prefetch next K-step's rows while we compute
        if (kk + KTILE < II && tid < KTILE) {
            __builtin_prefetch(bxk + (size_t)(KTILE + tid) * LL + (l0 > 0 ? l0 - PADW : 0), 0, 1);
        }
        __syncthreads();

        // ---- depthwise conv, store transposed f16 tile Xs[l][i] ----
        for (int idx = tid; idx < KTILE * LTILE; idx += 256) {
            const int il = idx & 31;
            const int ll = idx >> 5;
            float s = 0.0f;
            #pragma unroll
            for (int k = 0; k < KW; ++k)
                s = fmaf(cs[k][kk + il], Sraw[il][ll + k], s);
            Xs[ll][il] = (_Float16)s;
        }
        __syncthreads();

        // ---- A fragments from global bw (two 16B chunks per lane) ----
        v16h afrag[2];
        #pragma unroll
        for (int mt = 0; mt < 2; ++mt) {
            const _Float16* ap = bwb + (size_t)(wave * 32 + mt * 16 + mrow) * II + kk + khalf;
            const v8h lo = *(const v8h*)(ap);
            const v8h hi = *(const v8h*)(ap + 16);
            afrag[mt] = __builtin_shufflevector(lo, hi, 0, 1, 2, 3, 4, 5, 6, 7,
                                                        8, 9, 10, 11, 12, 13, 14, 15);
        }

        // ---- B fragments from LDS (two ds_load_b128 per lane) + 8 WMMAs ----
        #pragma unroll
        for (int nt = 0; nt < 4; ++nt) {
            const _Float16* xp = &Xs[nt * 16 + (lane & 15)][kbb];
            const v8h lo = *(const v8h*)(xp);
            const v8h hi = *(const v8h*)(xp + 16);
            const v16h bfrag = __builtin_shufflevector(lo, hi, 0, 1, 2, 3, 4, 5, 6, 7,
                                                               8, 9, 10, 11, 12, 13, 14, 15);
            #pragma unroll
            for (int mt = 0; mt < 2; ++mt)
                acc[mt][nt] = __builtin_amdgcn_wmma_f32_16x16x32_f16(
                    false, afrag[mt], false, bfrag, (short)0, acc[mt][nt], false, false);
        }
        __syncthreads();   // protect Xs/Sraw before next iteration's writes
    }

    // ---- epilogue: C layout M = j + 8*(lane>=16), N = lane&15 ----
    const int mh = (lane >> 4) * 8;
    const int nn = lane & 15;
    #pragma unroll
    for (int mt = 0; mt < 2; ++mt) {
        #pragma unroll
        for (int nt = 0; nt < 4; ++nt) {
            #pragma unroll
            for (int j = 0; j < 8; ++j) {
                const int m = wave * 32 + mt * 16 + mh + j;
                const int n = l0 + nt * 16 + nn;
                out[((size_t)(b * OO + m)) * LL + n] = acc[mt][nt][j] + pbs[m];
            }
        }
    }
}

extern "C" void kernel_launch(void* const* d_in, const int* in_sizes, int n_in,
                              void* d_out, int out_size, void* d_ws, size_t ws_size,
                              hipStream_t stream) {
    const float* bx     = (const float*)d_in[0];
    const float* t      = (const float*)d_in[1];
    const float* conv_w = (const float*)d_in[2];
    const float* proj_w = (const float*)d_in[3];
    const float* proj_b = (const float*)d_in[4];
    const float* mod_w  = (const float*)d_in[5];
    const float* mod_b  = (const float*)d_in[6];
    float* out = (float*)d_out;

    // workspace layout: [0, 2MB) bw f16 ; [2MB, 2MB+16KB) mod f32
    _Float16* bwbuf  = (_Float16*)d_ws;
    float*    modbuf = (float*)((char*)d_ws + (size_t)BB * OO * II * sizeof(_Float16));

    mod_kernel<<<BB, 256, 0, stream>>>(t, mod_w, mod_b, modbuf);
    bw_kernel<<<BB * OO, 256, 0, stream>>>(proj_w, modbuf, bwbuf);
    conv_gemm_kernel<<<BB * (LL / LTILE), 256, 0, stream>>>(bx, conv_w, proj_b, bwbuf, out);
}
